// NeoMindAttention_48610439856629
// MI455X (gfx1250) — matrix operand order repaired and checked by reading, not compiled, not run on hardware
//
#include <hip/hip_runtime.h>
#include <cstdint>
#include <cstddef>

// ---------------- types ----------------
typedef __bf16 bf16;
typedef __attribute__((ext_vector_type(16))) __bf16        v16bf;
typedef __attribute__((ext_vector_type(8)))  float         v8f;
typedef __attribute__((ext_vector_type(4)))  unsigned int  uint4v;
typedef __attribute__((ext_vector_type(4)))  int           v4i;

// CDNA5 async global->LDS path (guarded; falls back to sync staging)
#if __has_builtin(__builtin_amdgcn_global_load_async_to_lds_b128) && \
    __has_builtin(__builtin_amdgcn_s_wait_asynccnt)
#define NM_ASYNC 1
#else
#define NM_ASYNC 0
#endif

// ---------------- problem constants ----------------
static constexpr int Bb    = 2;
static constexpr int Tt    = 4096;
static constexpr int Cc    = 2048;
static constexpr int Hh    = 16;
static constexpr int Dh    = 128;
static constexpr int Mrows = Bb * Tt;          // 8192
static constexpr int W2    = 256;              // WINDOW/2
static constexpr int BQ    = 128;
static constexpr int NBLK  = Tt / BQ;          // 32
static constexpr int SPAN  = BQ + 2 * W2;      // 640
static constexpr int CH    = 32;               // key chunk per iteration
static constexpr int NCH   = SPAN / CH;        // 20

// ---------------- helpers ----------------
__device__ __forceinline__ unsigned short f32_to_bf16(float f) {
  union { float f; unsigned int u; } x; x.f = f;
  unsigned int u = x.u;
  if ((u & 0x7fffffffu) > 0x7f800000u) return (unsigned short)((u >> 16) | 0x0040u); // qNaN
  u += 0x7fffu + ((u >> 16) & 1u);   // round to nearest even
  return (unsigned short)(u >> 16);
}
__device__ __forceinline__ float bf16u_to_f32(unsigned short h) {
  union { unsigned int u; float f; } x; x.u = ((unsigned int)h) << 16; return x.f;
}
// build a 16-element bf16 fragment from two 8-element (16B) contiguous pieces
__device__ __forceinline__ v16bf ld_frag(const unsigned short* p0, const unsigned short* p1) {
  union { v16bf v; uint4v u[2]; } f;
  f.u[0] = *(const uint4v*)p0;
  f.u[1] = *(const uint4v*)p1;
  return f.v;
}
__device__ __forceinline__ v8f wmma_bf16(v16bf a, v16bf b, v8f c) {
  return __builtin_amdgcn_wmma_f32_16x16x32_bf16(false, a, false, b, (short)0, c, false, false);
}
__device__ __forceinline__ v8f zero_v8f() {
  v8f z;
#pragma unroll
  for (int i = 0; i < 8; ++i) z[i] = 0.0f;
  return z;
}
// async copy of 32B (two b128 transfers) global -> LDS, using IOFFSET for the 2nd half
__device__ __forceinline__ void async_copy32B(const unsigned short* g, unsigned short* l) {
#if NM_ASYNC
  __attribute__((address_space(1))) v4i* gp = (__attribute__((address_space(1))) v4i*)g;
  __attribute__((address_space(3))) v4i* lp = (__attribute__((address_space(3))) v4i*)l;
  __builtin_amdgcn_global_load_async_to_lds_b128(gp, lp, 0, 0);
  __builtin_amdgcn_global_load_async_to_lds_b128(gp, lp, 16, 0);
#else
  *(uint4v*)l       = *(const uint4v*)g;
  *(uint4v*)(l + 8) = *(const uint4v*)(g + 8);
#endif
}

// ---------------- kernel: fp32 -> bf16 convert ----------------
__global__ void nm_cvt_bf16(const float* __restrict__ src, unsigned short* __restrict__ dst, int n) {
  int i = (blockIdx.x * blockDim.x + threadIdx.x) * 4;
  if (i + 3 < n) {
    float4 v = *(const float4*)(src + i);
    dst[i + 0] = f32_to_bf16(v.x);
    dst[i + 1] = f32_to_bf16(v.y);
    dst[i + 2] = f32_to_bf16(v.z);
    dst[i + 3] = f32_to_bf16(v.w);
  } else {
    for (int j = i; j < n; ++j) dst[j] = f32_to_bf16(src[j]);
  }
}

// ---------------- kernel: transpose weight fp32[K][N] -> bf16[N][K] ----------------
__global__ void nm_wt_transpose(const float* __restrict__ W, unsigned short* __restrict__ Wt) {
  __shared__ float tile[32][33];
  int n0 = blockIdx.x * 32, k0 = blockIdx.y * 32;
  int c = threadIdx.x & 31, rbase = threadIdx.x >> 5;
#pragma unroll
  for (int i = 0; i < 4; ++i) {
    int r = rbase + i * 8;
    tile[r][c] = W[(size_t)(k0 + r) * Cc + (n0 + c)];
  }
  __syncthreads();
#pragma unroll
  for (int i = 0; i < 4; ++i) {
    int r = rbase + i * 8;
    Wt[(size_t)(n0 + r) * Cc + (k0 + c)] = f32_to_bf16(tile[c][r]);
  }
}

// ---------------- kernel: bf16 GEMM, Y[M][N] = A[M][K] * Wt[N][K]^T + bias ----------------
// 128x128 block tile, 8 waves in 4x2, each wave 32x64 (2x4 WMMA tiles), K-step 32.
// Double-buffered LDS; async global->LDS staging when available.
// layout_bhtd: write to (B,H,T,D) layout (bf16 obf and/or f32 of32); else f32 row-major [m][n].
__global__ __launch_bounds__(256) void nm_gemm_bf16(
    const unsigned short* __restrict__ A, const unsigned short* __restrict__ Wt,
    const float* __restrict__ bias,
    unsigned short* __restrict__ obf, float* __restrict__ of32, int layout_bhtd) {
  __shared__ unsigned short Ash[2][128 * 40];
  __shared__ unsigned short Bsh[2][128 * 40];

  const int tid  = threadIdx.x;
  const int lane = tid & 31;
  const int wv   = tid >> 5;
  const int wm   = wv >> 1;          // 0..3
  const int wn   = wv & 1;           // 0..1
  const bool lo  = lane < 16;
  const int l15  = lane & 15;

  const int ntile = Cc / 128;        // 16
  const int tm = blockIdx.x / ntile;
  const int tn = blockIdx.x % ntile;
  const int m0 = tm * 128, n0 = tn * 128;

  const int ldr = tid >> 1;          // 0..127
  const int ldk = (tid & 1) * 16;    // 0 or 16

  v8f acc[2][4];
#pragma unroll
  for (int i = 0; i < 2; ++i)
#pragma unroll
    for (int j = 0; j < 4; ++j) acc[i][j] = zero_v8f();

  const unsigned short* gA = A  + (size_t)(m0 + ldr) * Cc + ldk;
  const unsigned short* gB = Wt + (size_t)(n0 + ldr) * Cc + ldk;
  const int NK = Cc / 32;            // 64 K-steps

  auto issue = [&](int k0, int buf) {
    async_copy32B(gA + k0, &Ash[buf][ldr * 40 + ldk]);
    async_copy32B(gB + k0, &Bsh[buf][ldr * 40 + ldk]);
  };

  issue(0, 0);
  for (int it = 0; it < NK; ++it) {
    const int cur = it & 1;
    if (it + 1 < NK) {
      issue((it + 1) * 32, cur ^ 1);       // prefetch next tile into other buffer
#if NM_ASYNC
      __builtin_amdgcn_s_wait_asynccnt(4); // own 4 copies of stage `it` have landed
#endif
    } else {
#if NM_ASYNC
      __builtin_amdgcn_s_wait_asynccnt(0);
#endif
    }
    __syncthreads();                       // stage `it` visible to all waves

    v16bf af[2];
#pragma unroll
    for (int mt = 0; mt < 2; ++mt) {
      const unsigned short* p = &Ash[cur][(wm * 32 + mt * 16 + l15) * 40 + (lo ? 0 : 8)];
      af[mt] = ld_frag(p, p + 16);   // A layout: K {0..7,16..23} / {8..15,24..31}
    }
#pragma unroll
    for (int nt = 0; nt < 4; ++nt) {
      const unsigned short* p = &Bsh[cur][(wn * 64 + nt * 16 + l15) * 40 + (lo ? 0 : 8)];
      v16bf bfr = ld_frag(p, p + 16); // B layout: K {0..7,16..23} / {8..15,24..31}
#pragma unroll
      for (int mt = 0; mt < 2; ++mt) acc[mt][nt] = wmma_bf16(af[mt], bfr, acc[mt][nt]);
    }
    __syncthreads();                 // all reads of `cur` done before it is overwritten
  }

  // epilogue
#pragma unroll
  for (int mt = 0; mt < 2; ++mt) {
#pragma unroll
    for (int nt = 0; nt < 4; ++nt) {
      const int col = n0 + wn * 64 + nt * 16 + l15;
      const int rw0 = m0 + wm * 32 + mt * 16 + (lo ? 0 : 8);
      const float bv = bias[col];
#pragma unroll
      for (int r = 0; r < 8; ++r) {
        const int m = rw0 + r;
        const float val = acc[mt][nt][r] + bv;
        if (layout_bhtd) {
          const int b = m / Tt, t = m % Tt;
          const int h = col >> 7, d = col & 127;
          const size_t idx = (((size_t)(b * Hh + h)) * Tt + t) * Dh + d;
          if (obf)  obf[idx]  = f32_to_bf16(val);
          if (of32) of32[idx] = val;
        } else {
          of32[(size_t)m * Cc + col] = val;
        }
      }
    }
  }
}

// ---------------- kernel: RoPE (in-place on bf16 (B,H,T,D); optional fp32 copy-out) ----------------
__global__ void nm_rope(unsigned short* __restrict__ buf, float* __restrict__ out32, int n) {
  int idx = blockIdx.x * blockDim.x + threadIdx.x;
  if (idx >= n) return;                       // n = Bb*Hh*Tt*(Dh/2)
  const int i  = idx & 63;                    // pair index 0..63
  const int t  = (idx >> 6) & (Tt - 1);
  const int bh = idx >> 18;                   // 64*4096 = 2^18
  const float theta = __expf(-((float)(2 * i) / (float)Dh) * 9.210340371976184f); // ln(1e4)
  float s, c;
  __sincosf((float)t * theta, &s, &c);
  const size_t base = ((size_t)bh * Tt + t) * Dh + 2 * i;
  const float x1 = bf16u_to_f32(buf[base]);
  const float x2 = bf16u_to_f32(buf[base + 1]);
  const float o1 = x1 * c - x2 * s;
  const float o2 = x1 * s + x2 * c;
  buf[base]     = f32_to_bf16(o1);
  buf[base + 1] = f32_to_bf16(o2);
  if (out32) { out32[base] = o1; out32[base + 1] = o2; }
}

// ---------------- kernel: sliding-window flash attention ----------------
// grid: (B*H*NBLK) blocks of 256 threads. Each block: one (b,h) query tile of 128 rows.
// Wave w owns query rows w*16..w*16+15; Q pinned in VGPRs; ctx accum 16x128 fp32 in VGPRs.
__global__ __launch_bounds__(256) void nm_attn(
    const unsigned short* __restrict__ qbf, const unsigned short* __restrict__ kbf,
    const unsigned short* __restrict__ vbf, unsigned short* __restrict__ ctx) {
  __shared__ unsigned short Ksh[CH * 136];     // 32 keys x 128 d (pad to 136)
  __shared__ unsigned short Vtsh[Dh * 40];     // transposed V: 128 d x 32 s (pad to 40)
  __shared__ float          Ssh[8 * 16 * 32];  // per-wave score staging
  __shared__ unsigned short Psh[8 * 16 * 32];  // per-wave probabilities (bf16)
  __shared__ float          Csh[128];          // per-row rescale factor
  __shared__ float          Lsh[128];          // per-row softmax denom

  const int tid  = threadIdx.x;
  const int lane = tid & 31;
  const int wv   = tid >> 5;
  const bool lo  = lane < 16;
  const int l15  = lane & 15;

  const int blk = blockIdx.x & (NBLK - 1);
  const int bh  = blockIdx.x >> 5;             // NBLK == 32
  const size_t kvbase = (size_t)bh * Tt * Dh;

  // pin Q fragments (16 rows x 128 d) in registers
  const int trow = blk * BQ + wv * 16 + l15;
  v16bf qf[4];
#pragma unroll
  for (int kb = 0; kb < 4; ++kb) {
    const unsigned short* p = qbf + kvbase + (size_t)trow * Dh + kb * 32 + (lo ? 0 : 8);
    qf[kb] = ld_frag(p, p + 16);
  }

  v8f acc[8];
#pragma unroll
  for (int d = 0; d < 8; ++d) acc[d] = zero_v8f();

  float mrun = -1e30f, lrun = 0.0f;            // meaningful in lanes 0..15
  const int myrow = wv * 16 + l15;             // local query row when lo

  for (int chn = 0; chn < NCH; ++chn) {
    const int s0 = chn * CH;
    __syncthreads();                           // protect LDS reuse from previous iter
    // ---- stage K chunk (async) and transposed V chunk
    {
      const int s    = tid >> 3;               // 0..31
      const int dblk = (tid & 7) * 16;         // 0..112
      const int rk   = blk * BQ + s0 + s - W2;
      unsigned short* lk = &Ksh[s * 136 + dblk];
      if (rk >= 0 && rk < Tt) {
        async_copy32B(kbf + kvbase + (size_t)rk * Dh + dblk, lk);
        const unsigned short* gv = vbf + kvbase + (size_t)rk * Dh + dblk;
        unsigned short tmp[16];
        *(uint4v*)&tmp[0] = *(const uint4v*)gv;
        *(uint4v*)&tmp[8] = *(const uint4v*)(gv + 8);
#pragma unroll
        for (int j = 0; j < 16; ++j) Vtsh[(dblk + j) * 40 + s] = tmp[j];
      } else {
        uint4v z = {0u, 0u, 0u, 0u};
        *(uint4v*)lk       = z;
        *(uint4v*)(lk + 8) = z;
#pragma unroll
        for (int j = 0; j < 16; ++j) Vtsh[(dblk + j) * 40 + s] = 0;
      }
    }
#if NM_ASYNC
    __builtin_amdgcn_s_wait_asynccnt(0);
#endif
    __syncthreads();

    // ---- S = Q K^T  (16 rows x 32 keys per wave), stage to LDS scaled
#pragma unroll
    for (int st = 0; st < 2; ++st) {
      v8f sa = zero_v8f();
#pragma unroll
      for (int kb = 0; kb < 4; ++kb) {
        const unsigned short* p = &Ksh[(st * 16 + l15) * 136 + kb * 32 + (lo ? 0 : 8)];
        v16bf bfr = ld_frag(p, p + 16);
        sa = wmma_bf16(qf[kb], bfr, sa);
      }
      const int mrow = wv * 16 + (lo ? 0 : 8);
      const int scol = st * 16 + l15;
#pragma unroll
      for (int r = 0; r < 8; ++r)
        Ssh[(mrow + r) * 32 + scol] = sa[r] * 0.08838834764831845f; // 1/sqrt(128)
    }
    __syncthreads();

    // ---- online softmax (lanes 0..15 each own one query row)
    if (lo) {
      float sv[CH]; bool msk[CH];
      float mloc = -1e30f;
#pragma unroll
      for (int j = 0; j < CH; ++j) {
        const int s  = s0 + j;
        const int rk = blk * BQ + s - W2;
        const bool ok = (s >= myrow) && (s <= myrow + 2 * W2) && (rk >= 0) && (rk < Tt);
        msk[j] = ok;
        const float x = Ssh[(wv * 16 + l15) * 32 + j];
        sv[j] = x;
        if (ok && x > mloc) mloc = x;
      }
      const float mnew = fmaxf(mrun, mloc);
      const float cfac = __expf(mrun - mnew);  // both -1e30 -> exp(0)=1
      float psum = 0.0f;
#pragma unroll
      for (int j = 0; j < CH; ++j) {
        const float p = msk[j] ? __expf(sv[j] - mnew) : 0.0f;
        psum += p;
        Psh[(wv * 16 + l15) * 32 + j] = f32_to_bf16(p);
      }
      lrun = lrun * cfac + psum;
      mrun = mnew;
      Csh[wv * 16 + l15] = cfac;
    }
    __syncthreads();

    // ---- rescale ctx accum, then ctx += P * V
    float c8[8];
#pragma unroll
    for (int r = 0; r < 8; ++r) c8[r] = Csh[wv * 16 + (lo ? 0 : 8) + r];
#pragma unroll
    for (int d = 0; d < 8; ++d)
#pragma unroll
      for (int r = 0; r < 8; ++r) acc[d][r] = acc[d][r] * c8[r];

    const unsigned short* pp = &Psh[(wv * 16 + l15) * 32 + (lo ? 0 : 8)];
    const v16bf pf = ld_frag(pp, pp + 16);
#pragma unroll
    for (int dt = 0; dt < 8; ++dt) {
      const unsigned short* p = &Vtsh[(dt * 16 + l15) * 40 + (lo ? 0 : 8)];
      v16bf bfr = ld_frag(p, p + 16);
      acc[dt] = wmma_bf16(pf, bfr, acc[dt]);
    }
  }

  // ---- finalize: ctx /= l, write bf16 (B,T,H*D)
  if (lo) Lsh[wv * 16 + l15] = lrun;
  __syncthreads();
  float li[8];
#pragma unroll
  for (int r = 0; r < 8; ++r) li[r] = 1.0f / Lsh[wv * 16 + (lo ? 0 : 8) + r];
  const int b = bh / Hh, h = bh % Hh;
  const int t0 = blk * BQ + wv * 16 + (lo ? 0 : 8);
#pragma unroll
  for (int dt = 0; dt < 8; ++dt) {
    const int col = h * Dh + dt * 16 + l15;
#pragma unroll
    for (int r = 0; r < 8; ++r) {
      const int t = t0 + r;
      ctx[((size_t)(b * Tt + t)) * Cc + col] = f32_to_bf16(acc[dt][r] * li[r]);
    }
  }
}

// ---------------- host launcher ----------------
extern "C" void kernel_launch(void* const* d_in, const int* in_sizes, int n_in,
                              void* d_out, int out_size, void* d_ws, size_t ws_size,
                              hipStream_t stream) {
  (void)in_sizes; (void)n_in; (void)out_size; (void)ws_size;
  const float* x  = (const float*)d_in[0];
  const float* Wq = (const float*)d_in[1];
  const float* bq = (const float*)d_in[2];
  const float* Wk = (const float*)d_in[3];
  const float* bk = (const float*)d_in[4];
  const float* Wv = (const float*)d_in[5];
  const float* bv = (const float*)d_in[6];
  const float* Wo = (const float*)d_in[7];
  const float* bo = (const float*)d_in[8];

  float* out  = (float*)d_out;                              // (B,T,C)
  float* kout = out  + (size_t)Mrows * Cc;                  // (B,H,T,D)
  float* vout = kout + (size_t)Mrows * Cc;                  // (B,H,T,D)

  // workspace carve-up (~192 MB total; assumes ws_size is sufficient)
  char* w = (char*)d_ws;
  const size_t SZ_XB = (size_t)Mrows * Cc * 2;   // 32 MB bf16 activation-sized
  const size_t SZ_WT = (size_t)Cc * Cc * 2;      // 8 MB  bf16 weight-sized
  unsigned short* xbf  = (unsigned short*)(w);
  unsigned short* WqT  = (unsigned short*)(w + SZ_XB);
  unsigned short* WkT  = (unsigned short*)(w + SZ_XB + 1 * SZ_WT);
  unsigned short* WvT  = (unsigned short*)(w + SZ_XB + 2 * SZ_WT);
  unsigned short* WoT  = (unsigned short*)(w + SZ_XB + 3 * SZ_WT);
  unsigned short* qbf  = (unsigned short*)(w + SZ_XB + 4 * SZ_WT);
  unsigned short* kbf  = (unsigned short*)(w + 2 * SZ_XB + 4 * SZ_WT);
  unsigned short* vbf  = (unsigned short*)(w + 3 * SZ_XB + 4 * SZ_WT);
  unsigned short* ctxb = (unsigned short*)(w + 4 * SZ_XB + 4 * SZ_WT);

  // 1) convert x to bf16
  {
    const int n = Mrows * Cc;
    nm_cvt_bf16<<<(n / 4 + 255) / 256, 256, 0, stream>>>(x, xbf, n);
  }
  // 2) transpose+convert weights
  {
    dim3 g(Cc / 32, Cc / 32);
    nm_wt_transpose<<<g, 256, 0, stream>>>(Wq, WqT);
    nm_wt_transpose<<<g, 256, 0, stream>>>(Wk, WkT);
    nm_wt_transpose<<<g, 256, 0, stream>>>(Wv, WvT);
    nm_wt_transpose<<<g, 256, 0, stream>>>(Wo, WoT);
  }
  // 3) Q/K/V projections (bf16 WMMA GEMMs); V also writes fp32 to d_out
  {
    const int nblks = (Mrows / 128) * (Cc / 128);            // 1024
    nm_gemm_bf16<<<nblks, 256, 0, stream>>>(xbf, WqT, bq, qbf, nullptr, 1);
    nm_gemm_bf16<<<nblks, 256, 0, stream>>>(xbf, WkT, bk, kbf, nullptr, 1);
    nm_gemm_bf16<<<nblks, 256, 0, stream>>>(xbf, WvT, bv, vbf, vout,   1);
  }
  // 4) RoPE on q (in place) and k (in place + fp32 k output)
  {
    const int n = Bb * Hh * Tt * (Dh / 2);                   // 8388608
    nm_rope<<<(n + 255) / 256, 256, 0, stream>>>(qbf, nullptr, n);
    nm_rope<<<(n + 255) / 256, 256, 0, stream>>>(kbf, kout,   n);
  }
  // 5) sliding-window flash attention
  nm_attn<<<Bb * Hh * NBLK, 256, 0, stream>>>(qbf, kbf, vbf, ctxb);
  // 6) output projection -> fp32 out
  {
    const int nblks = (Mrows / 128) * (Cc / 128);
    nm_gemm_bf16<<<nblks, 256, 0, stream>>>(ctxb, WoT, bo, nullptr, out, 0);
  }
}
